// W2NER_14474039788085
// MI455X (gfx1250) — compile-verified, matrix operand order checked by reading, not run on hardware
//
#include <hip/hip_runtime.h>

#define B_      2
#define S_      192
#define H_      768
#define NH_     8
#define HD_     96
#define C_      128
#define CONVO_  384
#define NLNL_   49

typedef __attribute__((ext_vector_type(16))) __bf16 bf16x16;
typedef __attribute__((ext_vector_type(8)))  float  floatx8;

__device__ inline __bf16 f2b(float x) {
    union { float f; unsigned u; } a; a.f = x;
    unsigned r = a.u + 0x7FFFu + ((a.u >> 16) & 1u);   // RNE to bf16
    union { unsigned short s; __bf16 b; } o; o.s = (unsigned short)(r >> 16);
    return o.b;
}

// Low 32 bits of a flat shared-aperture pointer == LDS byte offset.
__device__ inline unsigned lds_off(const void* p) {
    return (unsigned)(size_t)p;
}

// Async global->LDS 16-byte copy (CDNA5, ASYNCcnt-tracked).
__device__ inline void async_copy16(unsigned lds_byte, const void* gsrc) {
    asm volatile("global_load_async_to_lds_b128 %0, %1, off"
                 :: "v"(lds_byte), "v"((unsigned long long)(size_t)gsrc)
                 : "memory");
}
__device__ inline void async_wait0() {
    asm volatile("s_wait_asynccnt 0" ::: "memory");
}

// Load a 16x32 bf16 fragment (A from [m][k] rows, B from [n][k] rows) from LDS.
// ISA 16-bit A/B layout: lane L -> row L%16; element v (pair h) -> k =
// 16*(v>=4) + 8*(L>=16) + 2*(v&3) + h.
__device__ inline bf16x16 load_frag(const __bf16* base, int ld) {
    int lane = threadIdx.x & 31;
    int m    = lane & 15;
    int hi   = lane >> 4;
    const __bf16* row = base + m * ld;
    bf16x16 f;
#pragma unroll
    for (int v = 0; v < 8; ++v) {
        int k = ((v >> 2) << 4) + (hi << 3) + ((v & 3) << 1);
        f[2 * v]     = row[k];
        f[2 * v + 1] = row[k + 1];
    }
    return f;
}

__device__ inline floatx8 wmma_bf16(bf16x16 a, bf16x16 b, floatx8 c) {
    return __builtin_amdgcn_wmma_f32_16x16x32_bf16(false, a, false, b,
                                                   (short)0, c, false, false);
}

// ---------------------------------------------------------------------------
// Generic GEMM: Y[M,N] = op(X[M,K] @ W[K,N] + bias + addsrc), fp32 I/O,
// bf16 WMMA compute. Tile 64x32, 128 threads (4 waves, one 16-row strip each).
// M%64==0, N%32==0, K%32==0 for all call sites.
// ---------------------------------------------------------------------------
__global__ __launch_bounds__(128)
void gemm_kernel(const float* __restrict__ X, int lda,
                 const float* __restrict__ W, int ldb,
                 const float* __restrict__ bias,
                 const float* __restrict__ addsrc, int ldadd,
                 float* __restrict__ Y, int ldc,
                 int M, int N, int K, int relu) {
    __shared__ __bf16 sA[64][32];   // [m][k]
    __shared__ __bf16 sB[32][32];   // [n][k]
    int m0 = blockIdx.x * 64;
    int n0 = blockIdx.y * 32;
    int tid = threadIdx.x, wave = tid >> 5, lane = tid & 31;

    floatx8 acc0 = {}, acc1 = {};
    for (int k0 = 0; k0 < K; k0 += 32) {
        __syncthreads();
        for (int i = tid; i < 64 * 32; i += 128) {
            int mm = i >> 5, kk = i & 31;
            sA[mm][kk] = f2b(X[(long)(m0 + mm) * lda + k0 + kk]);
        }
        for (int i = tid; i < 32 * 32; i += 128) {
            int nn = i >> 5, kk = i & 31;
            sB[nn][kk] = f2b(W[(long)(k0 + kk) * ldb + n0 + nn]);
        }
        if (k0 + 32 < K) {          // prefetch next k-chunk into L2/WGP$
            __builtin_prefetch(&X[(long)(m0 + (tid >> 1)) * lda + k0 + 32], 0, 3);
            __builtin_prefetch(&W[(long)(k0 + 32 + (tid & 31)) * ldb + n0], 0, 3);
        }
        __syncthreads();
        bf16x16 fa  = load_frag(&sA[wave * 16][0], 32);
        bf16x16 fb0 = load_frag(&sB[0][0],  32);
        bf16x16 fb1 = load_frag(&sB[16][0], 32);
        acc0 = wmma_bf16(fa, fb0, acc0);
        acc1 = wmma_bf16(fa, fb1, acc1);
    }
    int n_ = lane & 15;
    int mb = (lane >> 4) << 3;
#pragma unroll
    for (int r = 0; r < 8; ++r) {
        int m  = m0 + wave * 16 + mb + r;
        int c0 = n0 + n_, c1 = n0 + 16 + n_;
        float v0 = acc0[r], v1 = acc1[r];
        if (bias)   { v0 += bias[c0]; v1 += bias[c1]; }
        if (addsrc) { v0 += addsrc[(long)m * ldadd + c0];
                      v1 += addsrc[(long)m * ldadd + c1]; }
        if (relu)   { v0 = fmaxf(v0, 0.f); v1 = fmaxf(v1, 0.f); }
        Y[(long)m * ldc + c0] = v0;
        Y[(long)m * ldc + c1] = v1;
    }
}

// ---------------------------------------------------------------------------
// Attention scores + softmax. Block = (i-tile of 16 rows, head, batch).
// P[b,h,i,j] = softmax_j( mask(Q_i . K_j) / sqrt(96) ), fp32 to workspace.
// ---------------------------------------------------------------------------
__global__ __launch_bounds__(128)
void attn_scores_kernel(const float* __restrict__ Q,
                        const float* __restrict__ Km,
                        const int* __restrict__ mask,
                        float* __restrict__ P) {
    __shared__ __bf16 lq[16][HD_];     // 3 KB
    __shared__ __bf16 lk[S_][HD_];     // 36 KB  [n][k] == B-fragment layout
    __shared__ float  ls[16][S_];      // 12 KB
    int i0 = blockIdx.x * 16, h = blockIdx.y, b = blockIdx.z;
    int tid = threadIdx.x, wave = tid >> 5, lane = tid & 31;

    const float* qp = Q  + (long)b * S_ * H_ + h * HD_;
    const float* kp = Km + (long)b * S_ * H_ + h * HD_;
    for (int i = tid; i < 16 * HD_; i += 128) {
        int mm = i / HD_, kk = i % HD_;
        lq[mm][kk] = f2b(qp[(long)(i0 + mm) * H_ + kk]);
    }
    for (int i = tid; i < S_ * HD_; i += 128) {
        int nn = i / HD_, kk = i % HD_;
        lk[nn][kk] = f2b(kp[(long)nn * H_ + kk]);
    }
    __syncthreads();

    const float sc = 0.1020620726159658f;   // 1/sqrt(96)
    int n_ = lane & 15, mb = (lane >> 4) << 3;
#pragma unroll
    for (int tn = 0; tn < 3; ++tn) {        // each wave: cols wave*48 .. +47
        int nb = wave * 48 + tn * 16;
        floatx8 acc = {};
#pragma unroll
        for (int kc = 0; kc < 3; ++kc) {
            bf16x16 fa = load_frag(&lq[0][kc * 32],  HD_);
            bf16x16 fb = load_frag(&lk[nb][kc * 32], HD_);
            acc = wmma_bf16(fa, fb, acc);
        }
#pragma unroll
        for (int r = 0; r < 8; ++r) ls[mb + r][nb + n_] = acc[r] * sc;
    }
    __syncthreads();

    if (tid < 16) {                          // row-serial softmax (192 wide)
        float mx = -1e30f;
        for (int j = 0; j < S_; ++j) {
            float s = (mask[b * S_ + j] == 0) ? -1e9f : ls[tid][j];
            ls[tid][j] = s;
            mx = fmaxf(mx, s);
        }
        float sum = 0.f;
        for (int j = 0; j < S_; ++j) { float e = __expf(ls[tid][j] - mx); ls[tid][j] = e; sum += e; }
        float inv = 1.f / sum;
        for (int j = 0; j < S_; ++j) ls[tid][j] *= inv;
    }
    __syncthreads();
    float* pp = P + ((long)(b * NH_ + h) * S_ + i0) * S_;
    for (int i = tid; i < 16 * S_; i += 128) pp[i] = ls[i / S_][i % S_];
}

// ---------------------------------------------------------------------------
// 1D conv (per branch) as tap-decomposed WMMA GEMM:
//   y[b,t,c] = relu( sum_dt seq[b,t+dt,:] @ w[c,:,dt] + cb[c] )
// Tile 64 rows (over B*S) x 32 channels. Zero-pad rows at sequence edges.
// ---------------------------------------------------------------------------
__global__ __launch_bounds__(128)
void conv_kernel(const float* __restrict__ seq,     // [B*S, H]
                 const float* __restrict__ cw,      // [C, H, ksz]
                 const float* __restrict__ cb,      // [C]
                 float* __restrict__ out,           // [B*S, 384]
                 int ksz, int ncol0) {
    __shared__ __bf16 sA[64][32];
    __shared__ __bf16 sB[32][32];
    int m0 = blockIdx.x * 64;
    int n0 = blockIdx.y * 32;
    int tid = threadIdx.x, wave = tid >> 5, lane = tid & 31;
    int half = ksz >> 1;

    floatx8 acc0 = {}, acc1 = {};
    for (int tap = 0; tap < ksz; ++tap) {
        int dt = tap - half;
        for (int k0 = 0; k0 < H_; k0 += 32) {
            __syncthreads();
            for (int i = tid; i < 64 * 32; i += 128) {
                int mm = i >> 5, kk = i & 31;
                int r = m0 + mm, b = r / S_, t = r % S_ + dt;
                float v = (t >= 0 && t < S_)
                              ? seq[((long)b * S_ + t) * H_ + k0 + kk] : 0.f;
                sA[mm][kk] = f2b(v);
            }
            for (int i = tid; i < 32 * 32; i += 128) {
                int nn = i >> 5, kk = i & 31;
                sB[nn][kk] = f2b(cw[((long)(n0 + nn) * H_ + k0 + kk) * ksz + tap]);
            }
            __syncthreads();
            bf16x16 fa  = load_frag(&sA[wave * 16][0], 32);
            bf16x16 fb0 = load_frag(&sB[0][0],  32);
            bf16x16 fb1 = load_frag(&sB[16][0], 32);
            acc0 = wmma_bf16(fa, fb0, acc0);
            acc1 = wmma_bf16(fa, fb1, acc1);
        }
    }
    int n_ = lane & 15, mb = (lane >> 4) << 3;
#pragma unroll
    for (int r = 0; r < 8; ++r) {
        int m  = m0 + wave * 16 + mb + r;
        int c0 = n0 + n_, c1 = n0 + 16 + n_;
        float v0 = fmaxf(acc0[r] + cb[c0], 0.f);
        float v1 = fmaxf(acc1[r] + cb[c1], 0.f);
        out[(long)m * CONVO_ + ncol0 + c0] = v0;
        out[(long)m * CONVO_ + ncol0 + c1] = v1;
    }
}

// --------------------------- LayerNorm over 384 ----------------------------
__global__ __launch_bounds__(128)
void ln_kernel(const float* __restrict__ x, const float* __restrict__ g,
               const float* __restrict__ bta, float* __restrict__ y) {
    __shared__ float red[128];
    int row = blockIdx.x, tid = threadIdx.x;
    const float* xr = x + (long)row * CONVO_;
    float s = 0.f;
    for (int i = tid; i < CONVO_; i += 128) s += xr[i];
    red[tid] = s; __syncthreads();
    for (int st = 64; st > 0; st >>= 1) { if (tid < st) red[tid] += red[tid + st]; __syncthreads(); }
    float mean = red[0] / (float)CONVO_;
    __syncthreads();
    float v = 0.f;
    for (int i = tid; i < CONVO_; i += 128) { float d = xr[i] - mean; v += d * d; }
    red[tid] = v; __syncthreads();
    for (int st = 64; st > 0; st >>= 1) { if (tid < st) red[tid] += red[tid + st]; __syncthreads(); }
    float inv = rsqrtf(red[0] / (float)CONVO_ + 1e-5f);
    for (int i = tid; i < CONVO_; i += 128)
        y[(long)row * CONVO_ + i] = (xr[i] - mean) * inv * g[i] + bta[i];
}

// ---------------------------------------------------------------------------
// Pre-pack W2 and Wc to bf16 in k-chunked [n][k] tile order so the pair
// kernel can stage them with pure async block copies:
//   W2s[kc][nn][kk] = bf16(W2[(kc*32+kk)*384 + nn])   kc<24, nn<384, kk<32
//   Wcs[kc][nn][kk] = bf16(Wc[(kc*32+kk)*49  + nn])   kc<12, nn<64 (0-pad)
// ---------------------------------------------------------------------------
__global__ __launch_bounds__(256)
void pack_kernel(const float* __restrict__ W2, const float* __restrict__ Wc,
                 __bf16* __restrict__ W2s, __bf16* __restrict__ Wcs) {
    int tid = blockIdx.x * 256 + threadIdx.x;
    int stride = gridDim.x * 256;
    for (int i = tid; i < 24 * 384 * 32; i += stride) {
        int kc = i / (384 * 32), r = i % (384 * 32);
        int nn = r >> 5, kk = r & 31;
        W2s[i] = f2b(W2[(long)(kc * 32 + kk) * 384 + nn]);
    }
    for (int i = tid; i < 12 * 64 * 32; i += stride) {
        int kc = i / (64 * 32), r = i % (64 * 32);
        int nn = r >> 5, kk = r & 31;
        Wcs[i] = f2b(nn < NLNL_ ? Wc[(long)(kc * 32 + kk) * NLNL_ + nn] : 0.f);
    }
}

// ---------------------------------------------------------------------------
// Fused pair MLP. Block = (j-tile of 16, i, b).
//   h1 = relu(Ai[i] + Ci[i] + Bj[j] + Ci[j] + E[j-i+255])   (b1 folded in Ai)
//   h2 = relu(h1 @ W2 + b2)           768 -> 384
//   out = h2 @ Wc + bc                384 -> 49 (padded to 64)
// W2/Wc tiles staged bf16 via GLOBAL_LOAD_ASYNC_TO_LDS_B128 (ASYNCcnt).
// LDS: h1 24K | sW (W2 chunk, reused as h2) 24K | sWc chunk 4K | rowA 3K
// ---------------------------------------------------------------------------
__global__ __launch_bounds__(128)
void pair_mlp_kernel(const float* __restrict__ Ai, const float* __restrict__ Bj,
                     const float* __restrict__ Ci, const float* __restrict__ E,
                     const __bf16* __restrict__ W2s, const float* __restrict__ b2,
                     const __bf16* __restrict__ Wcs, const float* __restrict__ bc,
                     float* __restrict__ out) {
    __shared__ __bf16 h1[16][H_];        // 24 KB
    __shared__ __bf16 sW[384][32];       // 24 KB, later aliased as h2[16][384]
    __shared__ __bf16 sWc[64][32];       // 4 KB
    __shared__ float  rowA[H_];          // 3 KB

    int j0 = blockIdx.x * 16;
    int i  = blockIdx.y;
    int b  = blockIdx.z;
    int tid = threadIdx.x, wave = tid >> 5, lane = tid & 31;
    int ri = b * S_ + i;

    for (int n = tid; n < H_; n += 128)
        rowA[n] = Ai[(long)ri * H_ + n] + Ci[(long)ri * H_ + n];
    __syncthreads();

    for (int idx = tid; idx < 16 * H_; idx += 128) {
        int jj = idx / H_, n = idx % H_;
        int j  = j0 + jj, rj = b * S_ + j;
        float v = rowA[n] + Bj[(long)rj * H_ + n] + Ci[(long)rj * H_ + n]
                + E[(long)(j - i + 255) * H_ + n];
        h1[jj][n] = f2b(v > 0.f ? v : 0.f);
    }
    __syncthreads();

    unsigned ldsW  = lds_off(&sW[0][0]);
    unsigned ldsWc = lds_off(&sWc[0][0]);

    // ---- h2 = relu(h1 @ W2 + b2): M=16, N=384, K=768. Wave w -> cols w*96..
    floatx8 acc[6] = {};
    for (int k0 = 0; k0 < H_; k0 += 32) {
        __syncthreads();
        {   // async stage of the pre-packed 24KB chunk: 1536 x 16B
            const __bf16* src = W2s + (long)(k0 >> 5) * (384 * 32);
            for (int idx = tid; idx < (384 * 32) / 8; idx += 128)
                async_copy16(ldsW + idx * 16, src + idx * 8);
            async_wait0();
        }
        __syncthreads();
        bf16x16 fa = load_frag(&h1[0][k0], H_);
#pragma unroll
        for (int t = 0; t < 6; ++t) {
            bf16x16 fb = load_frag(&sW[wave * 96 + t * 16][0], 32);
            acc[t] = wmma_bf16(fa, fb, acc[t]);
        }
    }
    __syncthreads();
    __bf16* h2 = &sW[0][0];              // alias: [16][384] row-major
    {
        int n_ = lane & 15, mb = (lane >> 4) << 3;
#pragma unroll
        for (int t = 0; t < 6; ++t) {
            int ncol = wave * 96 + t * 16 + n_;
#pragma unroll
            for (int r = 0; r < 8; ++r) {
                float v = acc[t][r] + b2[ncol];
                h2[(mb + r) * 384 + ncol] = f2b(v > 0.f ? v : 0.f);
            }
        }
    }
    __syncthreads();

    // ---- out = h2 @ Wc + bc: M=16, K=384, N=64 (49 valid). Wave w -> n-tile w.
    floatx8 occ = {};
    for (int k0 = 0; k0 < 384; k0 += 32) {
        __syncthreads();
        {   // async stage of the 4KB Wc chunk: 256 x 16B
            const __bf16* src = Wcs + (long)(k0 >> 5) * (64 * 32);
            for (int idx = tid; idx < (64 * 32) / 8; idx += 128)
                async_copy16(ldsWc + idx * 16, src + idx * 8);
            async_wait0();
        }
        __syncthreads();
        bf16x16 fa = load_frag(h2 + k0, 384);
        bf16x16 fb = load_frag(&sWc[wave * 16][0], 32);
        occ = wmma_bf16(fa, fb, occ);
    }
    {
        int n_ = wave * 16 + (lane & 15);
        int mb = (lane >> 4) << 3;
        if (n_ < NLNL_) {
#pragma unroll
            for (int r = 0; r < 8; ++r) {
                int j = j0 + mb + r;
                out[(((long)b * S_ + i) * S_ + j) * NLNL_ + n_] = occ[r] + bc[n_];
            }
        }
    }
}

// ---------------------------------------------------------------------------
extern "C" void kernel_launch(void* const* d_in, const int* in_sizes, int n_in,
                              void* d_out, int out_size, void* d_ws, size_t ws_size,
                              hipStream_t stream) {
    const float* hs   = (const float*)d_in[0];
    const int*   am   = (const int*)  d_in[1];
    const float* Wq   = (const float*)d_in[2];  const float* bq = (const float*)d_in[3];
    const float* Wk   = (const float*)d_in[4];  const float* bk = (const float*)d_in[5];
    const float* Wv   = (const float*)d_in[6];  const float* bv = (const float*)d_in[7];
    const float* Wo   = (const float*)d_in[8];  const float* bo = (const float*)d_in[9];
    const float* cw3  = (const float*)d_in[10]; const float* cb3 = (const float*)d_in[11];
    const float* cw5  = (const float*)d_in[12]; const float* cb5 = (const float*)d_in[13];
    const float* cw7  = (const float*)d_in[14]; const float* cb7 = (const float*)d_in[15];
    const float* lng  = (const float*)d_in[16]; const float* lnb = (const float*)d_in[17];
    const float* dist = (const float*)d_in[18];
    const float* W1   = (const float*)d_in[19]; const float* b1 = (const float*)d_in[20];
    const float* W2   = (const float*)d_in[21]; const float* b2 = (const float*)d_in[22];
    const float* Wc   = (const float*)d_in[23]; const float* bc = (const float*)d_in[24];
    float* out = (float*)d_out;

    const long RH = (long)B_ * S_ * H_;        // 294912
    float* ws   = (float*)d_ws;
    float* q    = ws;
    float* kbuf = q    + RH;
    float* vbuf = kbuf + RH;
    float* ctx  = vbuf + RH;
    float* seq  = ctx  + RH;
    float* P    = seq  + RH;                   // B*NH*S*S = 589824
    float* craw = P    + (long)B_ * NH_ * S_ * S_;
    float* cln  = craw + (long)B_ * S_ * CONVO_;
    float* Ap   = cln  + (long)B_ * S_ * CONVO_;
    float* Bp   = Ap   + RH;
    float* Cp   = Bp   + RH;
    float* Em   = Cp   + RH;                   // 512*768
    __bf16* W2s = (__bf16*)(Em + (long)512 * H_);   // 24*384*32 bf16
    __bf16* Wcs = W2s + 24 * 384 * 32;              // 12*64*32 bf16

    dim3 blk(128);

    // pre-pack bf16 weight tiles for async staging in the pair kernel
    pack_kernel<<<dim3(64), dim3(256), 0, stream>>>(W2, Wc, W2s, Wcs);

    // Q/K/V projections: [384,768] x [768,768]
    gemm_kernel<<<dim3(6, 24), blk, 0, stream>>>(hs, H_, Wq, H_, bq, nullptr, 0, q,    H_, 384, H_, H_, 0);
    gemm_kernel<<<dim3(6, 24), blk, 0, stream>>>(hs, H_, Wk, H_, bk, nullptr, 0, kbuf, H_, 384, H_, H_, 0);
    gemm_kernel<<<dim3(6, 24), blk, 0, stream>>>(hs, H_, Wv, H_, bv, nullptr, 0, vbuf, H_, 384, H_, H_, 0);

    // scores + softmax -> P
    attn_scores_kernel<<<dim3(S_ / 16, NH_, B_), blk, 0, stream>>>(q, kbuf, am, P);

    // ctx[b,:,h*96:...] = P[b,h] @ V[b,:,h*96:...]   (192x96x192 per head)
    for (int b = 0; b < B_; ++b)
        for (int h = 0; h < NH_; ++h)
            gemm_kernel<<<dim3(3, 3), blk, 0, stream>>>(
                P + ((long)(b * NH_ + h)) * S_ * S_, S_,
                vbuf + (long)b * S_ * H_ + h * HD_, H_, nullptr, nullptr, 0,
                ctx + (long)b * S_ * H_ + h * HD_, H_, S_, HD_, S_, 0);

    // seq = hidden + ctx @ Wo + bo
    gemm_kernel<<<dim3(6, 24), blk, 0, stream>>>(ctx, H_, Wo, H_, bo, hs, H_, seq, H_, 384, H_, H_, 0);

    // conv branches -> craw[:, 0:128 | 128:256 | 256:384], then LayerNorm
    conv_kernel<<<dim3(6, 4), blk, 0, stream>>>(seq, cw3, cb3, craw, 3, 0);
    conv_kernel<<<dim3(6, 4), blk, 0, stream>>>(seq, cw5, cb5, craw, 5, 128);
    conv_kernel<<<dim3(6, 4), blk, 0, stream>>>(seq, cw7, cb7, craw, 7, 256);
    ln_kernel<<<dim3(B_ * S_), blk, 0, stream>>>(craw, lng, lnb, cln);

    // pair-factor projections (W1 split along its 1984 input rows)
    gemm_kernel<<<dim3(6, 24), blk, 0, stream>>>(seq, H_, W1,                 H_, b1, nullptr, 0, Ap, H_, 384, H_, H_, 0);
    gemm_kernel<<<dim3(6, 24), blk, 0, stream>>>(seq, H_, W1 + 768 * H_,      H_, nullptr, nullptr, 0, Bp, H_, 384, H_, H_, 0);
    gemm_kernel<<<dim3(6, 24), blk, 0, stream>>>(cln, CONVO_, W1 + 1536 * H_, H_, nullptr, nullptr, 0, Cp, H_, 384, H_, CONVO_, 0);
    gemm_kernel<<<dim3(8, 24), blk, 0, stream>>>(dist, 64, W1 + 1920 * H_,    H_, nullptr, nullptr, 0, Em, H_, 512, H_, 64, 0);

    // fused pair MLP -> out [B,S,S,49]
    pair_mlp_kernel<<<dim3(S_ / 16, S_, B_), blk, 0, stream>>>(Ap, Bp, Cp, Em, W2s, b2, Wcs, bc, out);
}
